// MotionEncoder_27857157881899
// MI455X (gfx1250) — matrix operand order repaired and checked by reading, not compiled, not run on hardware
//
#include <hip/hip_runtime.h>

typedef _Float16 h16;
typedef __attribute__((ext_vector_type(16))) _Float16 v16h;
typedef __attribute__((ext_vector_type(8)))  _Float16 v8h;
typedef __attribute__((ext_vector_type(8)))  float    v8f;
typedef __attribute__((ext_vector_type(4)))  float    v4f;

#define P_TOTAL   131072
#define IN_DIM    64
#define H_DIM     128
#define M_TILE    32           // points per workgroup (2 x 16-row WMMA tiles)
#define NTHREADS  256          // 8 wave32

// f16 weight workspace layout (offsets in halfs)
#define WIH0_OFF  0            // 512*64
#define WHH0_OFF  32768        // 512*128
#define WIH1_OFF  98304        // 512*128
#define WHH1_OFF  163840       // 512*128
#define WTOT      229376       // total halfs; fused biases (f32, 2*512) follow

// padded LDS row strides (halfs): +8 halfs (+4 dwords) staggers banks
#define XS_STRIDE 72           // IN_DIM + 8
#define HS_STRIDE 136          // H_DIM + 8

// ---------------------------------------------------------------- helpers
__device__ __forceinline__ v8f wmma16(v16h a, v16h b, v8f c) {
  // D = A(16x32 f16) x B(32x16 f16) + C(16x16 f32)
  return __builtin_amdgcn_wmma_f32_16x16x32_f16(
      /*neg_a=*/false, a, /*neg_b=*/false, b,
      /*c_mod=*/(short)0, c, /*reuse_a=*/false, /*reuse_b=*/false);
}

// A fragment from LDS (row-major, padded stride). Lane L: row = m0+(L&15),
// halfs [k0+16*(L>=16) .. +8) and [+8 .. +8) -> two ds_load_b128.
__device__ __forceinline__ v16h lds_A(const h16* base, int stride, int m0,
                                      int k0, int lane) {
  int m  = m0 + (lane & 15);
  int kk = k0 + ((lane >> 4) << 4);
  const h16* p = base + m * stride + kk;
  v8h lo = *(const v8h*)(p);
  v8h hi = *(const v8h*)(p + 8);
  v16h a;
#pragma unroll
  for (int i = 0; i < 8; ++i) { a[i] = lo[i]; a[i + 8] = hi[i]; }
  return a;
}

// B fragment from the (L2-resident) f16 copy of row-major W(512xK):
// lane L holds W[ncol, k0+16*(L>=16) .. +16) == W^T[k.., ncol].
// Regular (temporal) loads on purpose: weights are the reuse set.
__device__ __forceinline__ v16h glb_B(const h16* __restrict__ W, int K,
                                      int ncol, int k0, int lane) {
  const h16* p = W + (size_t)ncol * K + k0 + ((lane >> 4) << 4);
  return *(const v16h*)p;
}

__device__ __forceinline__ float sigf(float x) {
  float e = __expf(-x);
  return __builtin_amdgcn_rcpf(1.0f + e);
}
__device__ __forceinline__ float tanh_fast(float x) {
  float e = __expf(-2.0f * x);
  return (1.0f - e) * __builtin_amdgcn_rcpf(1.0f + e);
}
__device__ __forceinline__ v8f splat8(float b) {
  return (v8f){b, b, b, b, b, b, b, b};
}

// gate math + streaming (non-temporal) I/O for one 16x16 C/D tile
__device__ __forceinline__ void lstm_epilogue(
    const v8f* acc, int row0, int mt, int j, int lane, int layer,
    const float* __restrict__ c0, float* __restrict__ out,
    float* __restrict__ hn, float* __restrict__ cn, h16* s_dst) {
#pragma unroll
  for (int v = 0; v < 8; ++v) {
    int m = v + ((lane >> 4) << 3);        // C/D layout: M = v (+8 for hi half)
    int p = row0 + mt * 16 + m;
    size_t base = (size_t)p * (2 * H_DIM) + (size_t)layer * H_DIM + j;
    float ig = sigf(acc[0][v]);
    float fg = sigf(acc[1][v]);
    float gg = tanh_fast(acc[2][v]);
    float og = sigf(acc[3][v]);
    float cold = __builtin_nontemporal_load(&c0[base]);
    float cnew = fg * cold + ig * gg;
    float hnew = og * tanh_fast(cnew);
    __builtin_nontemporal_store(cnew, &cn[base]);
    __builtin_nontemporal_store(hnew, &hn[base]);
    if (s_dst) s_dst[(mt * 16 + m) * HS_STRIDE + j] = (h16)hnew;
    if (out) __builtin_nontemporal_store(hnew, &out[(size_t)p * H_DIM + j]);
  }
}

// ---------------------------------------------------------------- prep
// f32 -> f16 weight conversion + fused bias (b_ih + b_hh) into workspace.
__global__ __launch_bounds__(NTHREADS)
void lstm2_prep(const float* __restrict__ Wih0, const float* __restrict__ Whh0,
                const float* __restrict__ Wih1, const float* __restrict__ Whh1,
                const float* __restrict__ bih0, const float* __restrict__ bhh0,
                const float* __restrict__ bih1, const float* __restrict__ bhh1,
                h16* __restrict__ wh, float* __restrict__ btot) {
  int idx = blockIdx.x * NTHREADS + threadIdx.x;
  if (idx < 32768) {
    wh[WIH0_OFF + idx] = (h16)Wih0[idx];
  } else if (idx < 98304) {
    wh[idx] = (h16)Whh0[idx - 32768];
  } else if (idx < 163840) {
    wh[idx] = (h16)Wih1[idx - 98304];
  } else if (idx < 229376) {
    wh[idx] = (h16)Whh1[idx - 163840];
  } else if (idx < 229888) {
    int jj = idx - 229376;
    btot[jj] = bih0[jj] + bhh0[jj];
  } else if (idx < 230400) {
    int jj = idx - 229888;
    btot[512 + jj] = bih1[jj] + bhh1[jj];
  }
}

// ---------------------------------------------------------------- fused LSTM
__global__ __launch_bounds__(NTHREADS)
void lstm2_fused(const float* __restrict__ data,
                 const float* __restrict__ h0,
                 const float* __restrict__ c0,
                 const h16*   __restrict__ wh,
                 const float* __restrict__ btot,
                 float* __restrict__ out,
                 float* __restrict__ hn,
                 float* __restrict__ cn) {
  __shared__ __align__(16) h16 s_x  [M_TILE * XS_STRIDE];
  __shared__ __align__(16) h16 s_h00[M_TILE * HS_STRIDE];
  __shared__ __align__(16) h16 s_h01[M_TILE * HS_STRIDE];
  __shared__ __align__(16) h16 s_hl0[M_TILE * HS_STRIDE];

  const int tid  = threadIdx.x;
  const int lane = tid & 31;
  const int wave = tid >> 5;
  const int row0 = blockIdx.x * M_TILE;
  const int j    = (wave << 4) + (lane & 15);   // this lane's H column (0..127)

  // ---- stage x (f32 -> f16, NT reads), 16 float4 per row ----
  for (int q = tid; q < (M_TILE * IN_DIM) / 4; q += NTHREADS) {
    int r  = q >> 4;
    int c4 = (q & 15) << 2;
    v4f v = __builtin_nontemporal_load(
        (const v4f*)(data + (size_t)(row0 + r) * IN_DIM + c4));
    h16* d = &s_x[r * XS_STRIDE + c4];
    d[0] = (h16)v.x; d[1] = (h16)v.y; d[2] = (h16)v.z; d[3] = (h16)v.w;
  }
  // ---- stage h0 (both layers), 32 float4 per row per layer ----
  for (int q = tid; q < (M_TILE * H_DIM) / 4; q += NTHREADS) {
    int r  = q >> 5;
    int c4 = (q & 31) << 2;
    const float* src = h0 + (size_t)(row0 + r) * (2 * H_DIM);
    v4f v0 = __builtin_nontemporal_load((const v4f*)(src + c4));
    v4f v1 = __builtin_nontemporal_load((const v4f*)(src + H_DIM + c4));
    h16* d0 = &s_h00[r * HS_STRIDE + c4];
    h16* d1 = &s_h01[r * HS_STRIDE + c4];
    d0[0] = (h16)v0.x; d0[1] = (h16)v0.y; d0[2] = (h16)v0.z; d0[3] = (h16)v0.w;
    d1[0] = (h16)v1.x; d1[1] = (h16)v1.y; d1[2] = (h16)v1.z; d1[3] = (h16)v1.w;
  }
  __syncthreads();

  // =================== layer 0 ===================
  // Both M-tiles share every B fragment: 4 clause-able B loads feed 8 WMMAs.
  {
    v8f acc0[4], acc1[4];
#pragma unroll
    for (int g = 0; g < 4; ++g) {
      float b = btot[g * H_DIM + j];
      acc0[g] = splat8(b);
      acc1[g] = splat8(b);
    }

    // x @ W_ih0^T  (K = 64)
#pragma unroll
    for (int ks = 0; ks < IN_DIM / 32; ++ks) {
      v16h a0 = lds_A(s_x, XS_STRIDE, 0,  ks * 32, lane);
      v16h a1 = lds_A(s_x, XS_STRIDE, 16, ks * 32, lane);
      v16h b[4];
#pragma unroll
      for (int g = 0; g < 4; ++g)
        b[g] = glb_B(wh + WIH0_OFF, IN_DIM, g * H_DIM + j, ks * 32, lane);
#pragma unroll
      for (int g = 0; g < 4; ++g) {
        acc0[g] = wmma16(a0, b[g], acc0[g]);
        acc1[g] = wmma16(a1, b[g], acc1[g]);
      }
    }
    // h0[:,0] @ W_hh0^T  (K = 128)
#pragma unroll
    for (int ks = 0; ks < H_DIM / 32; ++ks) {
      v16h a0 = lds_A(s_h00, HS_STRIDE, 0,  ks * 32, lane);
      v16h a1 = lds_A(s_h00, HS_STRIDE, 16, ks * 32, lane);
      v16h b[4];
#pragma unroll
      for (int g = 0; g < 4; ++g)
        b[g] = glb_B(wh + WHH0_OFF, H_DIM, g * H_DIM + j, ks * 32, lane);
#pragma unroll
      for (int g = 0; g < 4; ++g) {
        acc0[g] = wmma16(a0, b[g], acc0[g]);
        acc1[g] = wmma16(a1, b[g], acc1[g]);
      }
    }
    lstm_epilogue(acc0, row0, 0, j, lane, 0, c0, nullptr, hn, cn, s_hl0);
    lstm_epilogue(acc1, row0, 1, j, lane, 0, c0, nullptr, hn, cn, s_hl0);
  }
  __syncthreads();

  // =================== layer 1 ===================
  {
    v8f acc0[4], acc1[4];
#pragma unroll
    for (int g = 0; g < 4; ++g) {
      float b = btot[512 + g * H_DIM + j];
      acc0[g] = splat8(b);
      acc1[g] = splat8(b);
    }

#pragma unroll
    for (int ks = 0; ks < H_DIM / 32; ++ks) {
      // h_l0 @ W_ih1^T
      {
        v16h a0 = lds_A(s_hl0, HS_STRIDE, 0,  ks * 32, lane);
        v16h a1 = lds_A(s_hl0, HS_STRIDE, 16, ks * 32, lane);
        v16h b[4];
#pragma unroll
        for (int g = 0; g < 4; ++g)
          b[g] = glb_B(wh + WIH1_OFF, H_DIM, g * H_DIM + j, ks * 32, lane);
#pragma unroll
        for (int g = 0; g < 4; ++g) {
          acc0[g] = wmma16(a0, b[g], acc0[g]);
          acc1[g] = wmma16(a1, b[g], acc1[g]);
        }
      }
      // h0[:,1] @ W_hh1^T
      {
        v16h a0 = lds_A(s_h01, HS_STRIDE, 0,  ks * 32, lane);
        v16h a1 = lds_A(s_h01, HS_STRIDE, 16, ks * 32, lane);
        v16h b[4];
#pragma unroll
        for (int g = 0; g < 4; ++g)
          b[g] = glb_B(wh + WHH1_OFF, H_DIM, g * H_DIM + j, ks * 32, lane);
#pragma unroll
        for (int g = 0; g < 4; ++g) {
          acc0[g] = wmma16(a0, b[g], acc0[g]);
          acc1[g] = wmma16(a1, b[g], acc1[g]);
        }
      }
    }
    lstm_epilogue(acc0, row0, 0, j, lane, 1, c0, out, hn, cn, nullptr);
    lstm_epilogue(acc1, row0, 1, j, lane, 1, c0, out, hn, cn, nullptr);
  }
}

// ---------------------------------------------------------------- launch
extern "C" void kernel_launch(void* const* d_in, const int* in_sizes, int n_in,
                              void* d_out, int out_size, void* d_ws, size_t ws_size,
                              hipStream_t stream) {
  const float* data  = (const float*)d_in[0];
  const float* h0    = (const float*)d_in[1];
  const float* c0    = (const float*)d_in[2];
  const float* W_ih0 = (const float*)d_in[3];
  const float* W_hh0 = (const float*)d_in[4];
  const float* b_ih0 = (const float*)d_in[5];
  const float* b_hh0 = (const float*)d_in[6];
  const float* W_ih1 = (const float*)d_in[7];
  const float* W_hh1 = (const float*)d_in[8];
  const float* b_ih1 = (const float*)d_in[9];
  const float* b_hh1 = (const float*)d_in[10];

  h16*   wh   = (h16*)d_ws;
  float* btot = (float*)((char*)d_ws + (size_t)WTOT * sizeof(h16));

  // outputs concatenated flat: output (P*128), hn (P*2*128), cn (P*2*128)
  float* out = (float*)d_out;
  float* hn  = out + (size_t)P_TOTAL * H_DIM;
  float* cn  = hn + (size_t)P_TOTAL * 2 * H_DIM;

  const int prep_items  = 230400;                       // weights + fused biases
  const int prep_blocks = (prep_items + NTHREADS - 1) / NTHREADS;
  lstm2_prep<<<prep_blocks, NTHREADS, 0, stream>>>(
      W_ih0, W_hh0, W_ih1, W_hh1, b_ih0, b_hh0, b_ih1, b_hh1, wh, btot);

  lstm2_fused<<<P_TOTAL / M_TILE, NTHREADS, 0, stream>>>(
      data, h0, c0, wh, btot, out, hn, cn);
}
